// RegisterTokenModule_74904229642715
// MI455X (gfx1250) — matrix operand order repaired and checked by reading, not compiled, not run on hardware
//
#include <hip/hip_runtime.h>
#include <hip/hip_bf16.h>
#include <cstdint>
#include <cstddef>

// ---------------------------------------------------------------------------
// Model dimensions (fixed by the reference)
// ---------------------------------------------------------------------------
#define BATCH   32
#define NCTX    576
#define DIN     768
#define DMODEL  512
#define NHEAD   8
#define NLAYER  4
#define NREG    32
#define NSHARED 8
#define DHEAD   64
#define DFF     2048
#define TSEQ    (NCTX + NREG)          // 608
#define MROWS   (BATCH * TSEQ)         // 19456  (divisible by 64)
#define MIN_ROWS (BATCH * NCTX)        // 18432  (divisible by 64)
#define D3      (3 * DMODEL)           // 1536

typedef __attribute__((ext_vector_type(16))) __bf16 v16bf;
typedef __attribute__((ext_vector_type(8)))  float  v8f;
typedef __attribute__((ext_vector_type(4)))  unsigned int u32x4;
typedef __attribute__((ext_vector_type(4)))  int    i32x4;
typedef __attribute__((ext_vector_type(8)))  int    i32x8;

// ---------------------------------------------------------------------------
// Tensor Data Mover availability / arity hedging.
//  - ROCm 7.2 (clang-22): 5-arg __builtin_amdgcn_tensor_load_to_lds
//  - amdgpu-toolchain (clang-23 + therock headers): 6-arg form (TDM header ships)
// ---------------------------------------------------------------------------
#if defined(__AMDGCN__) && __has_builtin(__builtin_amdgcn_tensor_load_to_lds)
#define ATH_TDM 1
#else
#define ATH_TDM 0
#endif

#if ATH_TDM
#if __has_include(<hip/amd_detail/amd_gfx1250_TDM.h>)
#define ATH_TDM_CALL(g0, g1, g2, g3) \
    __builtin_amdgcn_tensor_load_to_lds((g0), (g1), (g2), (g3), (i32x8)0, 0)
#else
#define ATH_TDM_CALL(g0, g1, g2, g3) \
    __builtin_amdgcn_tensor_load_to_lds((g0), (g1), (g2), (g3), 0)
#endif
#endif

// ---------------------------------------------------------------------------
// WMMA helper: D = A(16x32 bf16) x B(32x16 bf16) + C(16x16 f32)
// ---------------------------------------------------------------------------
__device__ __forceinline__ v8f wmma_bf16(v16bf a, v16bf b, v8f c) {
    return __builtin_amdgcn_wmma_f32_16x16x32_bf16(
        /*neg_a=*/false, a, /*neg_b=*/false, b,
        /*c_mod=*/(short)0, c, /*reuse_a=*/false, /*reuse_b=*/false);
}

// Load 8 consecutive bf16 (16 bytes) into fragment elements [base..base+7].
__device__ __forceinline__ void load8(const __bf16* p, v16bf& f, int base) {
    __bf16 tmp[8];
    *reinterpret_cast<uint4*>(tmp) = *reinterpret_cast<const uint4*>(p);
#pragma unroll
    for (int j = 0; j < 8; ++j) f[base + j] = tmp[j];
}

#if ATH_TDM
// ---------------------------------------------------------------------------
// Issue a TDM 2D tile load: global (row-major, row stride = stride_elems bf16)
// -> LDS at byte offset lds_off. tile_w contiguous bf16 per row, tile_h rows.
// D# layout per CDNA5 ISA ch.10.8 (group0 128b, group1 256b; groups 2/3 unused
// for 2D tiles). All operands are wave-uniform -> stay in SGPRs.
// ---------------------------------------------------------------------------
__device__ __forceinline__ void tdm_load_tile_bf16(const __bf16* gptr,
                                                   unsigned int lds_off,
                                                   int tile_w, int tile_h,
                                                   long stride_elems)
{
    const unsigned long long ga = (unsigned long long)(uintptr_t)gptr;
    u32x4 g0;
    g0[0] = 1u;                                        // count=1, user D#
    g0[1] = lds_off;                                   // lds_addr (bytes)
    g0[2] = (unsigned int)(ga & 0xFFFFFFFFu);          // global_addr[31:0]
    g0[3] = (unsigned int)((ga >> 32) & 0x01FFFFFFu)   // global_addr[56:32]
          | (2u << 30);                                // type = 2 ("image")
    const unsigned int td0 = 0x7FFFFFFFu;              // generous extents:
    const unsigned int td1 = 0x7FFFFFFFu;              // tiles are always in-bounds
    const unsigned long long s0 = (unsigned long long)stride_elems;
    i32x8 g1;
    g1[0] = (int)(1u << 16);                           // data_size=1 (2B), no mcast
    g1[1] = (int)((td0 & 0xFFFFu) << 16);              // [63:48] tensor_dim0 lo16
    g1[2] = (int)(((td0 >> 16) & 0xFFFFu) | ((td1 & 0xFFFFu) << 16));
    g1[3] = (int)(((td1 >> 16) & 0xFFFFu) | (((unsigned)tile_w & 0xFFFFu) << 16));
    g1[4] = (int)((unsigned)tile_h & 0xFFFFu);         // tile_dim1; tile_dim2=0
    g1[5] = (int)(s0 & 0xFFFFFFFFu);                   // tensor_dim0_stride lo32
    g1[6] = (int)((s0 >> 32) & 0xFFFFu);               // stride hi16; dim1_stride=0
    g1[7] = 0;
    const i32x4 z = (i32x4)0;                          // groups 2/3: 2D -> unused
    ATH_TDM_CALL(g0, g1, z, z);
}
#endif

// ---------------------------------------------------------------------------
// Generic tiled WMMA GEMM:
//   out = act( A_bf16[M,K] @ W_bf16[K,N] + bias ) (+ residual)
// Block: 256 threads (8 wave32). Tile: BM=64 x BN=128 x BK=32.
// Wave grid: 4 (M) x 2 (N); each wave owns a 16x64 slab = 4 accumulators.
// A tile: TDM double-buffered (async DMA overlapped with WMMA compute);
// B tile: manual transposed staging so B-fragments are contiguous ds_load_b128.
// ---------------------------------------------------------------------------
#define BM 64
#define BN 128
#define BKK 32
#define SA_BYTES (BM * BKK * 2)    // 4 KB per A buffer

template <bool RES, bool GELU, bool OBF16, bool REMAP>
__global__ __launch_bounds__(256)
void gemm_bf16_wmma(const __bf16* __restrict__ A,
                    const __bf16* __restrict__ W,
                    const float*  __restrict__ bias,
                    const float*  __restrict__ resid,
                    float*        __restrict__ outF,
                    __bf16*       __restrict__ outB,
                    int M, int N, int K, int rm_n, int rm_t)
{
    // NOTE: sA is declared FIRST so its static-LDS offset is 0; the TDM
    // descriptor addresses LDS by raw byte offset (buf0 @ 0, buf1 @ SA_BYTES).
    __shared__ __align__(16) __bf16 sA[2][BM][BKK];   // double-buffered A tile
    __shared__ __align__(16) __bf16 sBt[BN][BKK];     // B tile transposed [n][k]

    const int m0   = blockIdx.x * BM;
    const int n0   = blockIdx.y * BN;
    const int tid  = threadIdx.x;
    const int lane = tid & 31;
    const int wid  = tid >> 5;
    const int wrow = wid & 3;      // M sub-tile (16 rows each)
    const int wcol = wid >> 2;     // N sub-slab (64 cols each)
    const int fm   = lane & 15;    // fragment row / col index
    const int fg   = lane >> 4;    // lane half-group

    v8f acc[4] = {};

#if ATH_TDM
    // Prime the pipeline: DMA the k0=0 A tile into buffer 0 (wave 0 issues;
    // TDM ignores EXEC and is tracked with TENSORcnt).
    if (wid == 0)
        tdm_load_tile_bf16(&A[(size_t)m0 * K], 0u, BKK, BM, K);
#endif

    for (int k0 = 0; k0 < K; k0 += BKK) {
        const int buf = (k0 / BKK) & 1;

#if !ATH_TDM
        // ---- fallback: manual A staging (8 bf16 per thread, one b128) ----
        {
            const int r = tid >> 2;            // 0..63
            const int c = (tid & 3) * 8;       // 0,8,16,24
            *reinterpret_cast<uint4*>(&sA[buf][r][c]) =
                *reinterpret_cast<const uint4*>(&A[(size_t)(m0 + r) * K + k0 + c]);
        }
#endif
        // ---- stage B tile transposed: 4096 bf16, 16 per thread -----------
        {
            const int kk = tid >> 3;           // 0..31
            const int nn = (tid & 7) * 16;     // 0..112
            const __bf16* src = &W[(size_t)(k0 + kk) * N + n0 + nn];
            __bf16 tmp[16];
            *reinterpret_cast<uint4*>(tmp)     = *reinterpret_cast<const uint4*>(src);
            *reinterpret_cast<uint4*>(tmp + 8) = *reinterpret_cast<const uint4*>(src + 8);
#pragma unroll
            for (int j = 0; j < 16; ++j) sBt[nn + j][kk] = tmp[j];
        }

#if ATH_TDM
        // ---- issue next A-tile DMA, then wait for the current one --------
        // TDM ops from one wave complete in order, so TENSORcnt<=1 after
        // issuing the next tile means the current buffer has landed.
        if (wid == 0) {
            if (k0 + BKK < K) {
                tdm_load_tile_bf16(&A[(size_t)m0 * K + k0 + BKK],
                                   (unsigned)((buf ^ 1) * SA_BYTES), BKK, BM, K);
                __builtin_amdgcn_s_wait_tensorcnt(1);
            } else {
                __builtin_amdgcn_s_wait_tensorcnt(0);
            }
        }
#else
        if (k0 + BKK < K) {   // prefetch next tiles into cache
            __builtin_prefetch(&A[(size_t)(m0 + (tid >> 2)) * K + k0 + BKK], 0, 0);
            __builtin_prefetch(&W[(size_t)(k0 + BKK + (tid >> 3)) * N + n0 + (tid & 7) * 16], 0, 0);
        }
#endif
        __syncthreads();

        // ---- A fragment (16x32): lane fm = row; K 0..7/16..23 (fg=0),
        //      K 8..15/24..31 (fg=1) — contiguous ds_load_b128s ------------
        v16bf af;
        {
            const __bf16* pr = &sA[buf][wrow * 16 + fm][0];
            load8(pr + fg * 8,      af, 0);
            load8(pr + 16 + fg * 8, af, 8);
        }
        // ---- 4 B fragments (32x16 each) and 4 WMMAs ----------------------
#pragma unroll
        for (int t = 0; t < 4; ++t) {
            v16bf bfr;
            const int n  = wcol * 64 + t * 16 + fm;   // lane = column
            const int kb = fg * 8;                     // K 0..7/16..23 / 8..15/24..31
            load8(&sBt[n][kb],      bfr, 0);
            load8(&sBt[n][kb + 16], bfr, 8);
            acc[t] = wmma_bf16(af, bfr, acc[t]);
        }
        __syncthreads();
    }

    // ---- epilogue: C layout = row (fg*8 + r) x col (fm) -------------------
#pragma unroll
    for (int t = 0; t < 4; ++t) {
        const int n = n0 + wcol * 64 + t * 16 + fm;
        const float bv = bias[n];
#pragma unroll
        for (int r = 0; r < 8; ++r) {
            const int mrow = m0 + wrow * 16 + fg * 8 + r;
            float v = acc[t][r] + bv;
            if (GELU) v = 0.5f * v * (1.0f + erff(v * 0.70710678118654752f));
            size_t orow = (size_t)mrow;
            if (REMAP) orow = (size_t)(mrow / rm_n) * rm_t + (mrow % rm_n);
            const size_t idx = orow * (size_t)N + n;
            if (RES) v += resid[idx];
            if (OBF16) outB[idx] = (__bf16)v;
            else       outF[idx] = v;
        }
    }
}

// ---------------------------------------------------------------------------
// Flash attention, one wave32 per 16-query tile, 4 waves per block.
// qkv: bf16 [B*T, 3*D] (q | k | v), ctx: bf16 [B*T, D].
// Online softmax; scores & P.V done with v_wmma_f32_16x16x32_bf16.
// ---------------------------------------------------------------------------
__global__ __launch_bounds__(128)
void flash_attn_wmma(const __bf16* __restrict__ qkv, __bf16* __restrict__ ctx)
{
    __shared__ __align__(16) __bf16 sP[4][16][32];   // per-wave P scratch

    const int wid  = threadIdx.x >> 5;
    const int lane = threadIdx.x & 31;
    const int bh   = blockIdx.x;                 // 0..B*H-1
    const int b    = bh >> 3;
    const int h    = bh & 7;
    const int qt   = blockIdx.y * 4 + wid;       // query tile (16 rows)
    if (qt * 16 >= TSEQ) return;                 // wave-uniform exit

    const int fm = lane & 15;
    const int fg = lane >> 4;
    const int q0 = qt * 16;
    const size_t rowbase = (size_t)b * TSEQ;
    const int hoff = h * DHEAD;

    // Q fragments: per-lane row q0+fm, contiguous K chunks -> b128 loads
    v16bf aq[2];
#pragma unroll
    for (int c = 0; c < 2; ++c) {
        const __bf16* qp = &qkv[(rowbase + q0 + fm) * D3 + hoff + c * 32 + fg * 8];
        load8(qp,      aq[c], 0);
        load8(qp + 16, aq[c], 8);
    }

    float m_r[8], l_r[8];
#pragma unroll
    for (int r = 0; r < 8; ++r) { m_r[r] = -3.0e38f; l_r[r] = 0.0f; }
    v8f o[4] = {};

    const float scale = 0.125f;                  // 1/sqrt(64)

    for (int j = 0; j < TSEQ; j += 32) {
        // ---- scores: two 16x16 tiles (keys j..j+15, j+16..j+31) ----------
        v8f s0 = {}, s1 = {};
#pragma unroll
        for (int c = 0; c < 2; ++c) {
            // K^T fragment: lane = key, elements = contiguous features
            const __bf16* kp = &qkv[(rowbase + j + fm) * D3 + DMODEL + hoff + c * 32 + fg * 8];
            v16bf bk0, bk1;
            load8(kp,     bk0, 0);  load8(kp + 16, bk0, 8);
            s0 = wmma_bf16(aq[c], bk0, s0);
            const __bf16* kp2 = kp + (size_t)16 * D3;
            load8(kp2,    bk1, 0);  load8(kp2 + 16, bk1, 8);
            s1 = wmma_bf16(aq[c], bk1, s1);
        }

        // ---- scale + register-causal mask + online softmax ---------------
#pragma unroll
        for (int r = 0; r < 8; ++r) {
            const int qrow = q0 + fg * 8 + r;
            const int c0 = j + fm, c1 = c0 + 16;
            float x0 = s0[r] * scale;
            float x1 = s1[r] * scale;
            if (qrow >= NCTX && c0 > qrow) x0 = -3.0e38f;
            if (qrow >= NCTX && c1 > qrow) x1 = -3.0e38f;
            float mx = fmaxf(x0, x1);
            mx = fmaxf(mx, __shfl_xor(mx, 1));
            mx = fmaxf(mx, __shfl_xor(mx, 2));
            mx = fmaxf(mx, __shfl_xor(mx, 4));
            mx = fmaxf(mx, __shfl_xor(mx, 8));       // row max within 16-lane group
            const float mn = fmaxf(m_r[r], mx);
            const float a  = __expf(m_r[r] - mn);
            const float p0 = __expf(x0 - mn);
            const float p1 = __expf(x1 - mn);
            float ps = p0 + p1;
            ps += __shfl_xor(ps, 1);
            ps += __shfl_xor(ps, 2);
            ps += __shfl_xor(ps, 4);
            ps += __shfl_xor(ps, 8);                 // row sum
            l_r[r] = l_r[r] * a + ps;
            m_r[r] = mn;
#pragma unroll
            for (int t = 0; t < 4; ++t) o[t][r] *= a;   // rescale accum
            sP[wid][fg * 8 + r][fm]      = (__bf16)p0;  // stage P (bf16)
            sP[wid][fg * 8 + r][16 + fm] = (__bf16)p1;
        }
        // wave-local LDS producer->consumer: wait on DS counter (CDNA5)
        asm volatile("s_wait_dscnt 0x0" ::: "memory");

        // ---- P (16x32) @ V (32x64) --------------------------------------
        v16bf ap;
        {
            const __bf16* pp = &sP[wid][fm][0];
            load8(pp + fg * 8,      ap, 0);
            load8(pp + 16 + fg * 8, ap, 8);
        }
#pragma unroll
        for (int t = 0; t < 4; ++t) {
            v16bf bv;
#pragma unroll
            for (int i = 0; i < 16; ++i)     // lane = output feature, gather over keys
                bv[i] = qkv[(rowbase + j + fg * 8 + (i & 7) + (i >> 3) * 16) * D3 + 2 * DMODEL + hoff + t * 16 + fm];
            o[t] = wmma_bf16(ap, bv, o[t]);
        }
    }

    // ---- normalize and write ctx (bf16) ----------------------------------
#pragma unroll
    for (int t = 0; t < 4; ++t) {
#pragma unroll
        for (int r = 0; r < 8; ++r) {
            const int qrow = q0 + fg * 8 + r;
            const float v = o[t][r] / l_r[r];
            ctx[(rowbase + qrow) * DMODEL + hoff + t * 16 + fm] = (__bf16)v;
        }
    }
}

// ---------------------------------------------------------------------------
// LayerNorm over D=512, fp32 in -> bf16 out. One block (256 thr) per row.
// ---------------------------------------------------------------------------
__global__ __launch_bounds__(256)
void layernorm_to_bf16(const float* __restrict__ x,
                       const float* __restrict__ g,
                       const float* __restrict__ bta,
                       __bf16* __restrict__ out)
{
    const size_t row = blockIdx.x;
    const float* xr = x + row * DMODEL;
    const int t = threadIdx.x;
    const int lane = t & 31, wid = t >> 5;
    __shared__ float red[8];

    const float v0 = xr[t], v1 = xr[t + 256];
    float s = v0 + v1;
#pragma unroll
    for (int o = 16; o > 0; o >>= 1) s += __shfl_xor(s, o);
    if (lane == 0) red[wid] = s;
    __syncthreads();
    float tot = red[0];
#pragma unroll
    for (int i = 1; i < 8; ++i) tot += red[i];
    const float mu = tot * (1.0f / DMODEL);
    __syncthreads();

    const float d0 = v0 - mu, d1 = v1 - mu;
    float q = d0 * d0 + d1 * d1;
#pragma unroll
    for (int o = 16; o > 0; o >>= 1) q += __shfl_xor(q, o);
    if (lane == 0) red[wid] = q;
    __syncthreads();
    float qt = red[0];
#pragma unroll
    for (int i = 1; i < 8; ++i) qt += red[i];
    const float rs = rsqrtf(qt * (1.0f / DMODEL) + 1e-5f);

    out[row * DMODEL + t]       = (__bf16)(d0 * rs * g[t]       + bta[t]);
    out[row * DMODEL + t + 256] = (__bf16)(d1 * rs * g[t + 256] + bta[t + 256]);
}

// ---------------------------------------------------------------------------
// Final LayerNorm over register rows only -> d_out (shared | private | n_shared)
// ---------------------------------------------------------------------------
__global__ __launch_bounds__(256)
void final_layernorm_out(const float* __restrict__ x,
                         const float* __restrict__ g,
                         const float* __restrict__ bta,
                         float* __restrict__ out, int out_size)
{
    const int rowIdx = blockIdx.x;          // 0..B*NREG-1
    const int b  = rowIdx >> 5;
    const int ri = rowIdx & 31;
    const size_t row = (size_t)b * TSEQ + NCTX + ri;
    const float* xr = x + row * DMODEL;
    const int t = threadIdx.x;
    const int lane = t & 31, wid = t >> 5;
    __shared__ float red[8];

    const float v0 = xr[t], v1 = xr[t + 256];
    float s = v0 + v1;
#pragma unroll
    for (int o = 16; o > 0; o >>= 1) s += __shfl_xor(s, o);
    if (lane == 0) red[wid] = s;
    __syncthreads();
    float tot = red[0];
#pragma unroll
    for (int i = 1; i < 8; ++i) tot += red[i];
    const float mu = tot * (1.0f / DMODEL);
    __syncthreads();

    const float d0 = v0 - mu, d1 = v1 - mu;
    float q = d0 * d0 + d1 * d1;
#pragma unroll
    for (int o = 16; o > 0; o >>= 1) q += __shfl_xor(q, o);
    if (lane == 0) red[wid] = q;
    __syncthreads();
    float qt = red[0];
#pragma unroll
    for (int i = 1; i < 8; ++i) qt += red[i];
    const float rs = rsqrtf(qt * (1.0f / DMODEL) + 1e-5f);

    size_t dst;
    if (ri < NSHARED) dst = ((size_t)b * NSHARED + ri) * DMODEL;
    else dst = (size_t)BATCH * NSHARED * DMODEL
             + ((size_t)b * (NREG - NSHARED) + (ri - NSHARED)) * DMODEL;

    out[dst + t]       = d0 * rs * g[t]       + bta[t];
    out[dst + t + 256] = d1 * rs * g[t + 256] + bta[t + 256];

    if (rowIdx == 0 && t == 0 && out_size > BATCH * NREG * DMODEL)
        out[BATCH * NREG * DMODEL] = (float)NSHARED;   // trailing N_SHARED scalar
}

// ---------------------------------------------------------------------------
// Small utility kernels
// ---------------------------------------------------------------------------
__global__ void f32_to_bf16(const float* __restrict__ s, __bf16* __restrict__ d, long n) {
    long i  = (long)blockIdx.x * blockDim.x + threadIdx.x;
    long st = (long)gridDim.x * blockDim.x;
    for (; i < n; i += st) d[i] = (__bf16)s[i];
}

__global__ void init_registers(const float* __restrict__ rt, const float* __restrict__ rp,
                               const float* __restrict__ tt, float* __restrict__ x) {
    int i = blockIdx.x * blockDim.x + threadIdx.x;
    if (i >= BATCH * NREG * DMODEL) return;
    const int d = i % DMODEL;
    const int r = (i / DMODEL) % NREG;
    const int b = i / (DMODEL * NREG);
    const int rd = r * DMODEL + d;
    x[((size_t)b * TSEQ + NCTX + r) * DMODEL + d] = rt[rd] + rp[rd] + tt[rd];
}

// ---------------------------------------------------------------------------
// Host launcher
// ---------------------------------------------------------------------------
extern "C" void kernel_launch(void* const* d_in, const int* in_sizes, int n_in,
                              void* d_out, int out_size, void* d_ws, size_t ws_size,
                              hipStream_t stream) {
    const float* enc   = (const float*)d_in[0];
    const float* inW   = (const float*)d_in[1];
    const float* inB   = (const float*)d_in[2];
    const float* regT  = (const float*)d_in[3];
    const float* regP  = (const float*)d_in[4];
    const float* tokT  = (const float*)d_in[5];
    const float* ln1w  = (const float*)d_in[6];
    const float* ln1b  = (const float*)d_in[7];
    const float* Wqkv  = (const float*)d_in[8];
    const float* bqkv  = (const float*)d_in[9];
    const float* Wo    = (const float*)d_in[10];
    const float* bo    = (const float*)d_in[11];
    const float* ln2w  = (const float*)d_in[12];
    const float* ln2b  = (const float*)d_in[13];
    const float* W1    = (const float*)d_in[14];
    const float* b1    = (const float*)d_in[15];
    const float* W2    = (const float*)d_in[16];
    const float* b2    = (const float*)d_in[17];
    const float* lnfw  = (const float*)d_in[18];
    const float* lnfb  = (const float*)d_in[19];
    float* out = (float*)d_out;

    // ---- workspace layout (bump allocator, 256B aligned) -----------------
    char* p = (char*)d_ws;
    auto alloc = [&](size_t bytes) -> void* {
        void* r = (void*)p;
        p += (bytes + 255) & ~(size_t)255;
        return r;
    };
    __bf16* inw_bf  = (__bf16*)alloc((size_t)DIN * DMODEL * 2);
    __bf16* wqkv_bf = (__bf16*)alloc((size_t)NLAYER * DMODEL * D3 * 2);
    __bf16* wo_bf   = (__bf16*)alloc((size_t)NLAYER * DMODEL * DMODEL * 2);
    __bf16* w1_bf   = (__bf16*)alloc((size_t)NLAYER * DMODEL * DFF * 2);
    __bf16* w2_bf   = (__bf16*)alloc((size_t)NLAYER * DFF * DMODEL * 2);
    float*  x       = (float*) alloc((size_t)MROWS * DMODEL * 4);
    __bf16* xn_bf   = (__bf16*)alloc((size_t)MROWS * DMODEL * 2);
    __bf16* qkv_bf  = (__bf16*)alloc((size_t)MROWS * D3 * 2);
    __bf16* ctx_bf  = (__bf16*)alloc((size_t)MROWS * DMODEL * 2);
    __bf16* h_bf    = (__bf16*)alloc((size_t)MROWS * DFF * 2);   // also reused for enc_bf
    __bf16* enc_bf  = h_bf;   // alias: encoder bf16 only needed before layer 0

    const dim3 blk256(256);

    // ---- one-time-per-launch fp32 -> bf16 conversions --------------------
    auto conv = [&](const float* s, __bf16* d, long n) {
        int g = (int)((n + 255) / 256); if (g > 4096) g = 4096;
        hipLaunchKernelGGL(f32_to_bf16, dim3(g), blk256, 0, stream, s, d, n);
    };
    conv(enc,  enc_bf,  (long)BATCH * NCTX * DIN);
    conv(inW,  inw_bf,  (long)DIN * DMODEL);
    conv(Wqkv, wqkv_bf, (long)NLAYER * DMODEL * D3);
    conv(Wo,   wo_bf,   (long)NLAYER * DMODEL * DMODEL);
    conv(W1,   w1_bf,   (long)NLAYER * DMODEL * DFF);
    conv(W2,   w2_bf,   (long)NLAYER * DFF * DMODEL);

    // ---- input projection: x[:, :576, :] = enc @ inW + inB (row-remapped)
    hipLaunchKernelGGL((gemm_bf16_wmma<false, false, false, true>),
                       dim3(MIN_ROWS / BM, DMODEL / BN), blk256, 0, stream,
                       enc_bf, inw_bf, inB, (const float*)nullptr,
                       x, (__bf16*)nullptr, MIN_ROWS, DMODEL, DIN, NCTX, TSEQ);

    // ---- register tokens -> x[:, 576:, :] --------------------------------
    hipLaunchKernelGGL(init_registers,
                       dim3((BATCH * NREG * DMODEL + 255) / 256), blk256, 0, stream,
                       regT, regP, tokT, x);

    const int gM = MROWS / BM;   // 304
    for (int l = 0; l < NLAYER; ++l) {
        // LN1 -> xn (bf16)
        hipLaunchKernelGGL(layernorm_to_bf16, dim3(MROWS), blk256, 0, stream,
                           x, ln1w + (size_t)l * DMODEL, ln1b + (size_t)l * DMODEL, xn_bf);
        // QKV = xn @ Wqkv + bqkv  (bf16 out)
        hipLaunchKernelGGL((gemm_bf16_wmma<false, false, true, false>),
                           dim3(gM, D3 / BN), blk256, 0, stream,
                           xn_bf, wqkv_bf + (size_t)l * DMODEL * D3,
                           bqkv + (size_t)l * D3, (const float*)nullptr,
                           (float*)nullptr, qkv_bf, MROWS, D3, DMODEL, 1, 1);
        // attention -> ctx (bf16)
        hipLaunchKernelGGL(flash_attn_wmma,
                           dim3(BATCH * NHEAD, (TSEQ / 16 + 3) / 4), dim3(128), 0, stream,
                           qkv_bf, ctx_bf);
        // x += ctx @ Wo + bo
        hipLaunchKernelGGL((gemm_bf16_wmma<true, false, false, false>),
                           dim3(gM, DMODEL / BN), blk256, 0, stream,
                           ctx_bf, wo_bf + (size_t)l * DMODEL * DMODEL,
                           bo + (size_t)l * DMODEL, x,
                           x, (__bf16*)nullptr, MROWS, DMODEL, DMODEL, 1, 1);
        // LN2 -> xn
        hipLaunchKernelGGL(layernorm_to_bf16, dim3(MROWS), blk256, 0, stream,
                           x, ln2w + (size_t)l * DMODEL, ln2b + (size_t)l * DMODEL, xn_bf);
        // h = gelu(xn @ W1 + b1)  (bf16 out)
        hipLaunchKernelGGL((gemm_bf16_wmma<false, true, true, false>),
                           dim3(gM, DFF / BN), blk256, 0, stream,
                           xn_bf, w1_bf + (size_t)l * DMODEL * DFF,
                           b1 + (size_t)l * DFF, (const float*)nullptr,
                           (float*)nullptr, h_bf, MROWS, DFF, DMODEL, 1, 1);
        // x += h @ W2 + b2
        hipLaunchKernelGGL((gemm_bf16_wmma<true, false, false, false>),
                           dim3(gM, DMODEL / BN), blk256, 0, stream,
                           h_bf, w2_bf + (size_t)l * DFF * DMODEL,
                           b2 + (size_t)l * DMODEL, x,
                           x, (__bf16*)nullptr, MROWS, DMODEL, DFF, 1, 1);
    }

    // ---- final LN over register rows -> d_out ----------------------------
    hipLaunchKernelGGL(final_layernorm_out, dim3(BATCH * NREG), blk256, 0, stream,
                       x, lnfw, lnfb, out, out_size);
}
